// NonLocalBlock_35167192219677
// MI455X (gfx1250) — matrix operand order repaired and checked by reading, not compiled, not run on hardware
//
#include <hip/hip_runtime.h>

typedef __attribute__((ext_vector_type(16))) __bf16  v16bf;
typedef __attribute__((ext_vector_type(8)))  float   v8f;
typedef __attribute__((ext_vector_type(4)))  int     v4i;

#define B_  2
#define T_  7
#define C_  84      // 3*7*2*2
#define CP  96      // padded to 3 x K32 WMMA chunks (ch 84 of V = ones column)
#define HH  96
#define WHd 96
#define NN  (HH*WHd)   // 9216
#define H_  192
#define W_  192
#define KT  64      // keys per tile iteration
#define KPAD 104    // sK row pitch (bf16)
#define VPAD 72     // sV row pitch (bf16)
#define PPAD 68     // sP row pitch (f32)

// ---- optional gfx1250 async global->LDS path, with safe fallback ----------
#if defined(__has_builtin)
#  if __has_builtin(__builtin_amdgcn_global_load_async_to_lds_b128)
#    define USE_ASYNC_LDS 1
#  endif
#  if __has_builtin(__builtin_amdgcn_s_wait_asynccnt)
#    define WAIT_ASYNC() __builtin_amdgcn_s_wait_asynccnt(0)
#  endif
#endif
#ifndef WAIT_ASYNC
#  define WAIT_ASYNC() asm volatile("s_wait_asynccnt 0x0" ::: "memory")
#endif

typedef __attribute__((address_space(1))) v4i* g4p;   // global int4*
typedef __attribute__((address_space(3))) v4i* l4p;   // LDS int4*

__device__ __forceinline__ void cp_b128(const uint4* src, uint4* dst) {
#if defined(USE_ASYNC_LDS)
    v4i* s_gen = const_cast<v4i*>(reinterpret_cast<const v4i*>(src));
    v4i* d_gen = reinterpret_cast<v4i*>(dst);
    __builtin_amdgcn_global_load_async_to_lds_b128((g4p)s_gen, (l4p)d_gen, 0, 0);
#else
    *dst = *src;
#endif
}

// ---------------------------------------------------------------------------
// Kernel 1: pixel-unshuffle -> theta (f32, and bf16 zero-padded to CP)
// channel index = ((c*T + t)*2 + rh)*2 + rw ; n = hh*WH + wh
// ---------------------------------------------------------------------------
__global__ void k_theta(const float* __restrict__ xs,
                        float* __restrict__ th32,
                        __bf16* __restrict__ th16) {
    int idx = blockIdx.x * 256 + threadIdx.x;
    const int total = B_ * NN * CP;
    if (idx >= total) return;
    int ch = idx % CP;
    int n  = (idx / CP) % NN;
    int b  = idx / (CP * NN);
    float v = 0.0f;
    if (ch < C_) {
        int rw = ch & 1, rh = (ch >> 1) & 1, ct = ch >> 2;
        int t = ct % T_, c = ct / T_;
        int hh = n / WHd, wh = n % WHd;
        v = xs[(((size_t)(b*3 + c)*T_ + t)*H_ + (hh*2 + rh))*W_ + (wh*2 + rw)];
    }
    th32[idx] = v;
    th16[idx] = (__bf16)v;
}

// ---------------------------------------------------------------------------
// Kernel 2: g = theta @ Wg^T + bg, stored TRANSPOSED bf16 [B][CP][N].
// Channel C_ (84) is set to 1.0: the attention WMMA then accumulates the
// softmax row-sum "l" for free in O-tile 5 (ones-column trick).
// ---------------------------------------------------------------------------
__global__ void k_g(const float* __restrict__ th32,
                    const float* __restrict__ Wg,
                    const float* __restrict__ bg,
                    __bf16* __restrict__ gt) {
    int idx = blockIdx.x * 256 + threadIdx.x;
    const int total = B_ * NN * CP;
    if (idx >= total) return;
    int c = idx % CP;
    int n = (idx / CP) % NN;
    int b = idx / (CP * NN);
    float acc;
    if (c < C_) {
        acc = bg[c];
        const float* tr = th32 + ((size_t)b*NN + n) * CP;
        const float* wr = Wg + (size_t)c * C_;
        for (int k = 0; k < C_; ++k) acc += tr[k] * wr[k];
    } else {
        acc = (c == C_) ? 1.0f : 0.0f;   // ones column -> row-sum accumulator
    }
    gt[((size_t)b*CP + c)*NN + n] = (__bf16)acc;
}

// ---------------------------------------------------------------------------
// Kernel 3: flash-attention core.
// 4 waves/block, 16 query rows/wave. Double-buffered 64-key K/V tiles staged
// in LDS (async global->LDS when available). Per iteration per wave:
//   12 WMMA for S = Q*K^T, online max + exp, 12 WMMA for O += P*V.
// ---------------------------------------------------------------------------
__launch_bounds__(128)
__global__ void k_flash(const __bf16* __restrict__ th16,
                        const __bf16* __restrict__ gt,
                        float* __restrict__ y) {
    __shared__ __bf16 sK[2][KT][KPAD];   // keys x channels (row-major by key)
    __shared__ __bf16 sV[2][CP][VPAD];   // channels x keys (transposed g)
    __shared__ float  sP[4][16][PPAD];   // per-wave P staging (f32 -> bf16 A)

    const int tid  = threadIdx.x;
    const int lane = tid & 31;
    const int wave = tid >> 5;
    const int b    = blockIdx.y;
    const int row0 = (blockIdx.x * 4 + wave) * 16;
    const int lo   = lane & 15;
    const int hi   = lane >> 4;

    const __bf16* thb = th16 + (size_t)b * NN * CP;
    const __bf16* gtb = gt   + (size_t)b * CP * NN;

    // ---- load Q tile (16 x 96) into WMMA A-format: 3 chunks of 16x32 bf16
    v16bf qa[3];
    {
        const __bf16* qrow = thb + (size_t)(row0 + lo) * CP;
#pragma unroll
        for (int q = 0; q < 3; ++q) {
            int kb = q * 32 + hi * 8;
            union { v16bf v; uint4 u[2]; } un;
            un.u[0] = *(const uint4*)(qrow + kb);
            un.u[1] = *(const uint4*)(qrow + kb + 16);
            qa[q] = un.v;
        }
    }

    v8f o[6] = {};      // O: 16 rows x 96 cols; tile 5 col 4 carries row-sum l
    float m[8];
#pragma unroll
    for (int g = 0; g < 8; ++g) m[g] = -1e30f;

    // ---- cooperative tile fill: K (64x96) and V^T (96x64), 768 b128s each
    auto fill = [&](int buf, int j) {
        for (int i = tid; i < 768; i += 128) {
            int key = i / 12, part = i % 12;
            cp_b128((const uint4*)(thb + (size_t)(j + key)*CP + part*8),
                    (uint4*)&sK[buf][key][part*8]);
        }
        for (int i = tid; i < 768; i += 128) {
            int c = i / 8, part = i % 8;
            cp_b128((const uint4*)(gtb + (size_t)c*NN + j + part*8),
                    (uint4*)&sV[buf][c][part*8]);
        }
    };

    int buf = 0;
    fill(0, 0);
    WAIT_ASYNC();
    __syncthreads();

    for (int j = 0; j < NN; j += KT) {
        if (j + KT < NN) fill(buf ^ 1, j + KT);

        // ---- S = Q * K^T : four 16-key sub-tiles x 3 K-chunks
        v8f s[4];
#pragma unroll
        for (int blk = 0; blk < 4; ++blk) {
            const __bf16* krow = &sK[buf][blk*16 + lo][hi * 8];
            v8f acc = {};
#pragma unroll
            for (int q = 0; q < 3; ++q) {
                union { v16bf v; uint4 u[2]; } un;
                un.u[0] = *(const uint4*)(krow + q*32);
                un.u[1] = *(const uint4*)(krow + q*32 + 16);
                acc = __builtin_amdgcn_wmma_f32_16x16x32_bf16(
                        false, qa[q], false, un.v, (short)0, acc, false, false);
            }
            s[blk] = acc;
        }

        // ---- online softmax: row max (16-lane reduce), rescale O, P -> LDS
#pragma unroll
        for (int g = 0; g < 8; ++g) {
            float mv = fmaxf(fmaxf(s[0][g], s[1][g]), fmaxf(s[2][g], s[3][g]));
            mv = fmaxf(mv, __shfl_xor(mv, 1, 32));
            mv = fmaxf(mv, __shfl_xor(mv, 2, 32));
            mv = fmaxf(mv, __shfl_xor(mv, 4, 32));
            mv = fmaxf(mv, __shfl_xor(mv, 8, 32));
            float mn = fmaxf(m[g], mv);
            float sc = __expf(m[g] - mn);
            m[g] = mn;
#pragma unroll
            for (int t = 0; t < 6; ++t) o[t][g] *= sc;  // also rescales l col
            int r = hi * 8 + g;
#pragma unroll
            for (int blk = 0; blk < 4; ++blk)
                sP[wave][r][blk*16 + lo] = __expf(s[blk][g] - mn);
        }

        __builtin_amdgcn_wave_barrier();
        asm volatile("s_wait_dscnt 0x0" ::: "memory");

        // ---- P (16x64 f32) -> two bf16 A-format chunks via LDS transpose
        union { v16bf v; __bf16 e[16]; } pa[2];
#pragma unroll
        for (int a = 0; a < 2; ++a) {
            int base = a * 32 + hi * 8;
            float4 f0 = *(const float4*)&sP[wave][lo][base];
            float4 f1 = *(const float4*)&sP[wave][lo][base + 4];
            float4 f2 = *(const float4*)&sP[wave][lo][base + 16];
            float4 f3 = *(const float4*)&sP[wave][lo][base + 20];
            pa[a].e[0]  = (__bf16)f0.x; pa[a].e[1]  = (__bf16)f0.y;
            pa[a].e[2]  = (__bf16)f0.z; pa[a].e[3]  = (__bf16)f0.w;
            pa[a].e[4]  = (__bf16)f1.x; pa[a].e[5]  = (__bf16)f1.y;
            pa[a].e[6]  = (__bf16)f1.z; pa[a].e[7]  = (__bf16)f1.w;
            pa[a].e[8]  = (__bf16)f2.x; pa[a].e[9]  = (__bf16)f2.y;
            pa[a].e[10] = (__bf16)f2.z; pa[a].e[11] = (__bf16)f2.w;
            pa[a].e[12] = (__bf16)f3.x; pa[a].e[13] = (__bf16)f3.y;
            pa[a].e[14] = (__bf16)f3.z; pa[a].e[15] = (__bf16)f3.w;
        }
        __builtin_amdgcn_wave_barrier();

        // ---- O += P * V : 6 channel tiles x 2 K-chunks
#pragma unroll
        for (int t = 0; t < 6; ++t) {
#pragma unroll
            for (int a = 0; a < 2; ++a) {
                const __bf16* vrow = &sV[buf][t*16 + lo][a*32 + hi*8];
                union { v16bf v; uint4 u[2]; } un;
                un.u[0] = *(const uint4*)(vrow);
                un.u[1] = *(const uint4*)(vrow + 16);
                o[t] = __builtin_amdgcn_wmma_f32_16x16x32_bf16(
                         false, pa[a].v, false, un.v, (short)0, o[t], false, false);
            }
        }

        WAIT_ASYNC();
        __syncthreads();
        buf ^= 1;
    }

    // ---- extract l from the ones column (tile 5, col 4), normalize, write y
#pragma unroll
    for (int g = 0; g < 8; ++g) {
        float lsum = __shfl(o[5][g], (lane & 16) | 4, 32);
        float inv  = 1.0f / lsum;
        int r = row0 + hi * 8 + g;
#pragma unroll
        for (int t = 0; t < 6; ++t)
            y[((size_t)b*NN + r)*CP + t*16 + lo] = o[t][g] * inv;
    }
}

// ---------------------------------------------------------------------------
// Kernel 4: out = xs + pixel_shuffle(y @ Ww^T + bw)
// ---------------------------------------------------------------------------
__global__ void k_out(const float* __restrict__ xs,
                      const float* __restrict__ y,
                      const float* __restrict__ Ww,
                      const float* __restrict__ bw,
                      float* __restrict__ out) {
    int idx = blockIdx.x * 256 + threadIdx.x;
    const int total = B_ * NN * C_;
    if (idx >= total) return;
    int c = idx % C_;
    int n = (idx / C_) % NN;
    int b = idx / (C_ * NN);
    const float* yr = y + ((size_t)b*NN + n) * CP;
    const float* wr = Ww + (size_t)c * C_;
    float acc = bw[c];
    for (int k = 0; k < C_; ++k) acc += yr[k] * wr[k];
    int rw = c & 1, rh = (c >> 1) & 1, ct = c >> 2;
    int t = ct % T_, cc = ct / T_;
    int hh = n / WHd, wh = n % WHd;
    size_t oi = (((size_t)(b*3 + cc)*T_ + t)*H_ + (hh*2 + rh))*W_ + (wh*2 + rw);
    out[oi] = xs[oi] + acc;
}

// ---------------------------------------------------------------------------
extern "C" void kernel_launch(void* const* d_in, const int* in_sizes, int n_in,
                              void* d_out, int out_size, void* d_ws, size_t ws_size,
                              hipStream_t stream) {
    const float* xs = (const float*)d_in[0];
    const float* Wg = (const float*)d_in[1];
    const float* bg = (const float*)d_in[2];
    const float* Ww = (const float*)d_in[3];
    const float* bw = (const float*)d_in[4];
    float* out = (float*)d_out;

    char* ws = (char*)d_ws;
    const size_t nTh = (size_t)B_ * NN * CP;
    float*  th32 = (float*)ws;                               // nTh f32
    __bf16* th16 = (__bf16*)(ws + nTh*4);                    // nTh bf16
    __bf16* gt   = (__bf16*)(ws + nTh*4 + nTh*2);            // nTh bf16 (transposed)
    float*  yws  = (float*)(ws + nTh*4 + nTh*2 + nTh*2);     // nTh f32

    const int tot1 = B_ * NN * CP;
    k_theta<<<(tot1 + 255)/256, 256, 0, stream>>>(xs, th32, th16);
    k_g    <<<(tot1 + 255)/256, 256, 0, stream>>>(th32, Wg, bg, gt);

    dim3 gflash(NN / 64, B_);   // 4 waves/block, 16 rows/wave -> 64 rows/block
    k_flash<<<gflash, 128, 0, stream>>>(th16, gt, yws);

    const int tot4 = B_ * NN * C_;
    k_out<<<(tot4 + 255)/256, 256, 0, stream>>>(xs, yws, Ww, bw, out);
}
